// AllSetTransformer_g_28166395527445
// MI455X (gfx1250) — matrix operand order repaired and checked by reading, not compiled, not run on hardware
//
#include <hip/hip_runtime.h>

// ---------------- Problem constants (match reference) ----------------
#define Dm    256      // feature dim
#define Hh    8        // heads
#define DHd   32       // head dim
#define CHD   128      // classifier hidden
#define NCLS  10       // classes
#define MEDGE 20000    // hyperedges
#define BGR   64       // graphs
#define NEGS  0.2f     // leaky-relu slope

typedef __attribute__((ext_vector_type(16))) __bf16          v16bf;
typedef __attribute__((ext_vector_type(8)))  float           v8f;
typedef __attribute__((ext_vector_type(8)))  unsigned short  us8;

__device__ __forceinline__ unsigned short f2bf(float f) {
  unsigned int u = __float_as_uint(f);
  unsigned int r = (u + 0x7FFFu + ((u >> 16) & 1u)) >> 16;   // RNE
  return (unsigned short)r;
}

// -------------------------------------------------------------------
// elementwise helpers
// -------------------------------------------------------------------
__global__ void cvt_bf_kernel(const float* __restrict__ s, unsigned short* __restrict__ d, long n) {
  long i = (long)blockIdx.x * 256 + threadIdx.x;
  if (i < n) d[i] = f2bf(s[i]);
}

// convert + transpose a [K x ncols] f32 weight into [ncols x K] bf16
__global__ void cvt_bf_t_kernel(const float* __restrict__ W, unsigned short* __restrict__ WT,
                                int K, int ncols) {
  long i = (long)blockIdx.x * 256 + threadIdx.x;
  if (i >= (long)K * ncols) return;
  int f = (int)(i / ncols), c = (int)(i % ncols);
  WT[(long)c * K + f] = f2bf(W[i]);
}

__global__ void fillf_kernel(float* __restrict__ p, float v, long n) {
  long i = (long)blockIdx.x * 256 + threadIdx.x;
  if (i < n) p[i] = v;
}

// -------------------------------------------------------------------
// WMMA bf16 GEMM:  C[nrows,ncols] = act(A[nrows,256] @ B[256,ncols] + bias)
// B is given TRANSPOSED: BT[ncols][256] so all fragments load as
// contiguous 16B runs -> global_load_b128 straight into WMMA operands.
// block = 256 threads = 8 waves; wave -> 16x32 tile (one A fragment
// shared by two B column tiles => 3 b128 loads per WMMA);
// block tile 64x64.  ncols must be a multiple of 64 (256 or 128 here).
// ACT: 0 = none, 1 = relu.  Cf (f32) and/or Cb (bf16) outputs.
// -------------------------------------------------------------------
union FragBF { v16bf v; us8 h[2]; };

template <int ACT>
__global__ void gemm_bf16_wmma(const unsigned short* __restrict__ A,
                               const unsigned short* __restrict__ BT,
                               const float* __restrict__ bias,
                               float* __restrict__ Cf,
                               unsigned short* __restrict__ Cb,
                               int nrows, int ncols) {
  const int lane = threadIdx.x & 31;
  const int wave = threadIdx.x >> 5;
  const int half = lane >> 4;        // 0: lanes 0-15, 1: lanes 16-31
  const int l16  = lane & 15;
  const int r0 = blockIdx.x * 64 + (wave >> 1) * 16;
  const int c0 = blockIdx.y * 64 + (wave & 1) * 32;

  int arow = r0 + l16; if (arow >= nrows) arow = nrows - 1;   // clamp; guarded on store
  const unsigned short* __restrict__ ar  = A  + (long)arow * Dm;
  const unsigned short* __restrict__ bc0 = BT + (long)(c0 + l16) * Dm;
  const unsigned short* __restrict__ bc1 = bc0 + 16 * Dm;

  v8f acc0 = {}, acc1 = {};
#pragma unroll
  for (int kt = 0; kt < Dm; kt += 32) {
    FragBF a, b0, b1;
    __builtin_prefetch(ar + kt + 64, 0, 1);
    // A k order used here (contiguous, same as B): lane half h holds K = h*16 .. h*16+15 for its row
    a.h[0] = *(const us8*)(ar + kt + half * 16);
    a.h[1] = *(const us8*)(ar + kt + half * 16 + 8);
    // B k order used here (contiguous, same as A): lane half h holds K = h*16 .. h*16+15 for its column
    b0.h[0] = *(const us8*)(bc0 + kt + half * 16);
    b0.h[1] = *(const us8*)(bc0 + kt + half * 16 + 8);
    b1.h[0] = *(const us8*)(bc1 + kt + half * 16);
    b1.h[1] = *(const us8*)(bc1 + kt + half * 16 + 8);
    acc0 = __builtin_amdgcn_wmma_f32_16x16x32_bf16(false, a.v, false, b0.v,
                                                   (short)0, acc0, false, false);
    acc1 = __builtin_amdgcn_wmma_f32_16x16x32_bf16(false, a.v, false, b1.v,
                                                   (short)0, acc1, false, false);
  }

  const int  col0 = c0 + l16, col1 = col0 + 16;
  const float bcv0 = bias ? bias[col0] : 0.0f;
  const float bcv1 = bias ? bias[col1] : 0.0f;
#pragma unroll
  for (int v = 0; v < 8; ++v) {
    int orow = r0 + half * 8 + v;    // C/D layout: lanes0-15 M=v, lanes16-31 M=8+v
    if (orow < nrows) {
      float v0 = acc0[v] + bcv0;
      float v1 = acc1[v] + bcv1;
      if (ACT == 1) { v0 = v0 > 0.f ? v0 : 0.f; v1 = v1 > 0.f ? v1 : 0.f; }
      long off = (long)orow * ncols;
      if (Cf) { Cf[off + col0] = v0; Cf[off + col1] = v1; }
      if (Cb) { Cb[off + col0] = f2bf(v0); Cb[off + col1] = f2bf(v1); }
    }
  }
}

// -------------------------------------------------------------------
// fold att into Wk:  WkAtt[f,h] = sum_d Wk[f, h*32+d]*att[h,d];  bkAtt[h] likewise
// -------------------------------------------------------------------
__global__ void build_watt_kernel(const float* __restrict__ Wkl, const float* __restrict__ bkl,
                                  const float* __restrict__ attl,
                                  float* __restrict__ WkAtt, float* __restrict__ bkAtt) {
  int i = blockIdx.x * 256 + threadIdx.x;
  if (i < Dm * Hh) {
    int f = i >> 3, h = i & 7;
    float acc = 0.f;
    for (int dd = 0; dd < DHd; ++dd) acc += Wkl[(long)f * Dm + h * DHd + dd] * attl[h * DHd + dd];
    WkAtt[i] = acc;
  }
  if (i < Hh) {
    float acc = 0.f;
    for (int dd = 0; dd < DHd; ++dd) acc += bkl[i * DHd + dd] * attl[i * DHd + dd];
    bkAtt[i] = acc;
  }
}

// alpha[n,h] = x[n,:] @ WkAtt[:,h] + bkAtt[h]
__global__ void alpha_kernel(const float* __restrict__ x, const float* __restrict__ WkAtt,
                             const float* __restrict__ bkAtt, float* __restrict__ alpha, long n) {
  long i = (long)blockIdx.x * 256 + threadIdx.x;
  if (i >= n * Hh) return;
  long r = i >> 3; int h = (int)(i & 7);
  const float* xr = x + r * Dm;
  float acc = bkAtt[h];
  for (int f = 0; f < Dm; ++f) acc += xr[f] * WkAtt[f * Hh + h];
  alpha[i] = acc;
}

// -------------------------------------------------------------------
// segment softmax, scatter style
// -------------------------------------------------------------------
__global__ void edge_max_kernel(const float* __restrict__ alpha, const int* __restrict__ gidx,
                                const int* __restrict__ sidx, float* __restrict__ smax, long EH) {
  long i = (long)blockIdx.x * 256 + threadIdx.x;
  if (i >= EH) return;
  long e = i >> 3; int h = (int)(i & 7);
  float a = alpha[(long)gidx[e] * Hh + h];
  a = a > 0.f ? a : NEGS * a;                       // leaky relu
  atomicMax(&smax[(long)sidx[e] * Hh + h], a);      // global_atomic_max_num_f32
}

__global__ void edge_scatter_kernel(const float* __restrict__ alpha, const float* __restrict__ V,
                                    const int* __restrict__ gidx, const int* __restrict__ sidx,
                                    const float* __restrict__ smax, float* __restrict__ ssum,
                                    float* __restrict__ pooled, long Ecnt) {
  long i = (long)blockIdx.x * 256 + threadIdx.x;
  if (i >= Ecnt * Dm) return;
  long e = i >> 8; int d = (int)(i & 255); int h = d >> 5;
  int g = gidx[e], s = sidx[e];
  float a = alpha[(long)g * Hh + h];
  a = a > 0.f ? a : NEGS * a;
  float m = smax[(long)s * Hh + h];
  if (!(m > -1e29f)) m = 0.f;                       // empty-group: isfinite guard
  float ev = __expf(a - m);
  if ((d & 31) == 0) atomicAdd(&ssum[(long)s * Hh + h], ev);
  atomicAdd(&pooled[(long)s * Dm + d], ev * V[(long)g * Dm + d]);
}

// -------------------------------------------------------------------
// block-wide sum (blockDim multiple of 32, <= 256), wave32 shuffles + LDS
// -------------------------------------------------------------------
__device__ __forceinline__ float blockSum(float v, float* sm) {
#pragma unroll
  for (int o = 16; o > 0; o >>= 1) v += __shfl_xor(v, o, 32);
  const int w = threadIdx.x >> 5, nw = blockDim.x >> 5;
  __syncthreads();
  if ((threadIdx.x & 31) == 0) sm[w] = v;
  __syncthreads();
  if (threadIdx.x == 0) { float t = 0.f; for (int i = 0; i < nw; ++i) t += sm[i]; sm[0] = t; }
  __syncthreads();
  return sm[0];
}

// pooled/ssum + att residual + LayerNorm(ln0) -> u (f32) + ub (bf16)
__global__ void pma_post_kernel(const float* __restrict__ pooled, const float* __restrict__ ssum,
                                const float* __restrict__ attl, const float* __restrict__ g0,
                                const float* __restrict__ b0, float* __restrict__ u,
                                unsigned short* __restrict__ ub) {
  __shared__ float sm[8];
  const int r = blockIdx.x, d = threadIdx.x, h = d >> 5;
  float s = ssum[(long)r * Hh + h]; s = s > 0.f ? s : 1.f;
  float val = pooled[(long)r * Dm + d] / s + attl[d];
  float mu  = blockSum(val, sm) * (1.0f / Dm);
  float dv  = val - mu;
  float var = blockSum(dv * dv, sm) * (1.0f / Dm);
  float o = dv * rsqrtf(var + 1e-5f) * g0[d] + b0[d];
  u[(long)r * Dm + d]  = o;
  ub[(long)r * Dm + d] = f2bf(o);
}

// relu(LayerNorm(u + relu_rff, ln1)) -> cur
__global__ void pma_final_kernel(const float* __restrict__ u, const float* __restrict__ rff,
                                 const float* __restrict__ g1, const float* __restrict__ b1,
                                 float* __restrict__ outx) {
  __shared__ float sm[8];
  const int r = blockIdx.x, d = threadIdx.x;
  float val = u[(long)r * Dm + d] + rff[(long)r * Dm + d];
  float mu  = blockSum(val, sm) * (1.0f / Dm);
  float dv  = val - mu;
  float var = blockSum(dv * dv, sm) * (1.0f / Dm);
  float o = dv * rsqrtf(var + 1e-5f) * g1[d] + b1[d];
  outx[(long)r * Dm + d] = o > 0.f ? o : 0.f;
}

// -------------------------------------------------------------------
// readout + classifier
// -------------------------------------------------------------------
__global__ void readout_acc_kernel(const float* __restrict__ x, const int* __restrict__ batch,
                                   float* __restrict__ sums, float* __restrict__ cnt, long n) {
  long i = (long)blockIdx.x * 256 + threadIdx.x;
  if (i >= n * Dm) return;
  long r = i >> 8; int d = (int)(i & 255);
  int b = batch[r];
  atomicAdd(&sums[(long)b * Dm + d], x[r * Dm + d]);
  if (d == 0) atomicAdd(&cnt[b], 1.f);
}

__global__ void readout_div_kernel(const float* __restrict__ sums, const float* __restrict__ cnt,
                                   unsigned short* __restrict__ rb) {
  int i = blockIdx.x * 256 + threadIdx.x;
  if (i >= BGR * Dm) return;
  float c = cnt[i >> 8]; c = c > 0.f ? c : 1.f;
  rb[i] = f2bf(sums[i] / c);
}

__global__ void ln_relu_128_kernel(float* __restrict__ h, const float* __restrict__ g,
                                   const float* __restrict__ b) {
  __shared__ float sm[4];
  const int r = blockIdx.x, d = threadIdx.x;
  float val = h[(long)r * CHD + d];
  float mu  = blockSum(val, sm) * (1.0f / CHD);
  float dv  = val - mu;
  float var = blockSum(dv * dv, sm) * (1.0f / CHD);
  float o = dv * rsqrtf(var + 1e-5f) * g[d] + b[d];
  h[(long)r * CHD + d] = o > 0.f ? o : 0.f;
}

__global__ void cls_out_kernel(const float* __restrict__ h1, const float* __restrict__ W2c,
                               const float* __restrict__ b2c, float* __restrict__ out) {
  int i = blockIdx.x * 256 + threadIdx.x;
  if (i >= BGR * NCLS) return;
  int b = i / NCLS, c = i % NCLS;
  float acc = b2c[c];
  for (int k = 0; k < CHD; ++k) acc += h1[b * CHD + k] * W2c[k * NCLS + c];
  out[i] = acc;
}

// -------------------------------------------------------------------
// host orchestration
// -------------------------------------------------------------------
static inline unsigned gblk(long n) { return (unsigned)((n + 255) / 256); }

extern "C" void kernel_launch(void* const* d_in, const int* in_sizes, int n_in,
                              void* d_out, int out_size, void* d_ws, size_t ws_size,
                              hipStream_t stream) {
  const float* X        = (const float*)d_in[0];
  const int*   v2e_src  = (const int*)d_in[1];
  const int*   v2e_dst  = (const int*)d_in[2];
  const int*   all_batch= (const int*)d_in[3];
  const float* Wk  = (const float*)d_in[6];
  const float* bk  = (const float*)d_in[7];
  const float* Wv  = (const float*)d_in[8];
  const float* bv  = (const float*)d_in[9];
  const float* att = (const float*)d_in[10];
  const float* ln0g= (const float*)d_in[11];
  const float* ln0b= (const float*)d_in[12];
  const float* ln1g= (const float*)d_in[13];
  const float* ln1b= (const float*)d_in[14];
  const float* W1  = (const float*)d_in[15];
  const float* b1  = (const float*)d_in[16];
  const float* W2  = (const float*)d_in[17];
  const float* b2  = (const float*)d_in[18];
  const float* cW1 = (const float*)d_in[19];
  const float* cb1 = (const float*)d_in[20];
  const float* clng= (const float*)d_in[21];
  const float* clnb= (const float*)d_in[22];
  const float* cW2 = (const float*)d_in[23];
  const float* cb2 = (const float*)d_in[24];

  const int  N    = in_sizes[0] / Dm;
  const long Ecnt = in_sizes[1];
  const long NB   = (long)N * Dm;

  // ---- carve workspace ----
  char* p = (char*)d_ws;
  auto alloc = [&](size_t bytes) { void* q = p; p += (bytes + 255) & ~(size_t)255; return q; };
  float*          cur    = (float*)alloc(NB * 4);
  unsigned short* xb     = (unsigned short*)alloc(NB * 2);
  float*          Vb     = (float*)alloc(NB * 4);
  float*          pooled = (float*)alloc(NB * 4);   // reused as rFF output
  float*          u      = (float*)alloc(NB * 4);
  unsigned short* ub     = (unsigned short*)alloc(NB * 2);
  unsigned short* tb     = (unsigned short*)alloc(NB * 2);
  float*          alpha  = (float*)alloc((long)N * Hh * 4);
  float*          smax   = (float*)alloc((long)N * Hh * 4);
  float*          ssum   = (float*)alloc((long)N * Hh * 4);
  unsigned short* WvT    = (unsigned short*)alloc(4 * Dm * Dm * 2);  // [ncols][K] bf16
  unsigned short* W1T    = (unsigned short*)alloc(4 * Dm * Dm * 2);
  unsigned short* W2T    = (unsigned short*)alloc(4 * Dm * Dm * 2);
  unsigned short* cW1T   = (unsigned short*)alloc(CHD * Dm * 2);
  float*          WkAtt  = (float*)alloc(Dm * Hh * 4);
  float*          bkAtt  = (float*)alloc(Hh * 4);
  float*          sums   = (float*)alloc(BGR * Dm * 4);
  float*          cnt    = (float*)alloc(BGR * 4);
  unsigned short* rb     = (unsigned short*)alloc(BGR * Dm * 2);
  float*          h1     = (float*)alloc(BGR * CHD * 4);

  // ---- bf16 transposed weight conversion (once per call) ----
  for (int l = 0; l < 4; ++l) {
    const long o = (long)l * Dm * Dm;
    cvt_bf_t_kernel<<<gblk((long)Dm * Dm), 256, 0, stream>>>(Wv + o, WvT + o, Dm, Dm);
    cvt_bf_t_kernel<<<gblk((long)Dm * Dm), 256, 0, stream>>>(W1 + o, W1T + o, Dm, Dm);
    cvt_bf_t_kernel<<<gblk((long)Dm * Dm), 256, 0, stream>>>(W2 + o, W2T + o, Dm, Dm);
  }
  cvt_bf_t_kernel<<<gblk((long)Dm * CHD), 256, 0, stream>>>(cW1, cW1T, Dm, CHD);

  const int  n_src[4]  = {N, MEDGE, N, MEDGE};
  const int  n_seg[4]  = {MEDGE, N, MEDGE, N};
  const int* gi[4]     = {v2e_src, v2e_dst, v2e_src, v2e_dst};
  const int* si[4]     = {v2e_dst, v2e_src, v2e_dst, v2e_src};

  for (int l = 0; l < 4; ++l) {
    const float* xin = (l == 0) ? X : cur;
    const int n = n_src[l], ns = n_seg[l];
    const long woff = (long)l * Dm * Dm;

    cvt_bf_kernel<<<gblk((long)n * Dm), 256, 0, stream>>>(xin, xb, (long)n * Dm);

    // V = x @ Wv + bv  (WMMA)
    gemm_bf16_wmma<0><<<dim3((n + 63) / 64, Dm / 64), 256, 0, stream>>>(
        xb, WvT + woff, bv + l * Dm, Vb, nullptr, n, Dm);

    // attention scores (att folded into Wk)
    build_watt_kernel<<<gblk(Dm * Hh), 256, 0, stream>>>(
        Wk + woff, bk + l * Dm, att + l * Dm, WkAtt, bkAtt);
    alpha_kernel<<<gblk((long)n * Hh), 256, 0, stream>>>(xin, WkAtt, bkAtt, alpha, n);

    // segment softmax + weighted scatter of V
    fillf_kernel<<<gblk((long)ns * Hh), 256, 0, stream>>>(smax, -1e30f, (long)ns * Hh);
    fillf_kernel<<<gblk((long)ns * Hh), 256, 0, stream>>>(ssum, 0.f, (long)ns * Hh);
    fillf_kernel<<<gblk((long)ns * Dm), 256, 0, stream>>>(pooled, 0.f, (long)ns * Dm);
    edge_max_kernel<<<gblk(Ecnt * Hh), 256, 0, stream>>>(alpha, gi[l], si[l], smax, Ecnt * Hh);
    edge_scatter_kernel<<<gblk(Ecnt * Dm), 256, 0, stream>>>(
        alpha, Vb, gi[l], si[l], smax, ssum, pooled, Ecnt);

    // normalize + seed residual + LN0
    pma_post_kernel<<<ns, Dm, 0, stream>>>(pooled, ssum, att + l * Dm,
                                           ln0g + l * Dm, ln0b + l * Dm, u, ub);

    // rFF: t = relu(u@W1+b1) [bf16]; r = relu(t@W2+b2) [f32 into pooled]
    gemm_bf16_wmma<1><<<dim3((ns + 63) / 64, Dm / 64), 256, 0, stream>>>(
        ub, W1T + woff, b1 + l * Dm, nullptr, tb, ns, Dm);
    gemm_bf16_wmma<1><<<dim3((ns + 63) / 64, Dm / 64), 256, 0, stream>>>(
        tb, W2T + woff, b2 + l * Dm, pooled, nullptr, ns, Dm);

    // cur = relu(LN1(u + r))
    pma_final_kernel<<<ns, Dm, 0, stream>>>(u, pooled, ln1g + l * Dm, ln1b + l * Dm, cur);
  }

  // ---- grouped-mean readout + classifier ----
  fillf_kernel<<<gblk((long)BGR * Dm), 256, 0, stream>>>(sums, 0.f, (long)BGR * Dm);
  fillf_kernel<<<gblk(BGR), 256, 0, stream>>>(cnt, 0.f, BGR);
  readout_acc_kernel<<<gblk((long)N * Dm), 256, 0, stream>>>(cur, all_batch, sums, cnt, N);
  readout_div_kernel<<<gblk((long)BGR * Dm), 256, 0, stream>>>(sums, cnt, rb);

  gemm_bf16_wmma<0><<<dim3(1, CHD / 64), 256, 0, stream>>>(
      rb, cW1T, cb1, h1, nullptr, BGR, CHD);
  ln_relu_128_kernel<<<BGR, CHD, 0, stream>>>(h1, clng, clnb);
  cls_out_kernel<<<gblk(BGR * NCLS), 256, 0, stream>>>(h1, cW2, cb2, (float*)d_out);
}
